// BahdanauAttention_33981781246336
// MI455X (gfx1250) — matrix operand order repaired and checked by reading, not compiled, not run on hardware
//
#include <hip/hip_runtime.h>
#include <hip/hip_bf16.h>

// ---------------------------------------------------------------------------
// Bahdanau attention, MI455X (gfx1250, wave32, WMMA).
//   score GEMM: (B*P=50176, K=2048) x (2048, 512) in bf16 WMMA 16x16x32,
//   fused +att2 / ReLU / dot(w_full) epilogue, per-batch softmax, then
//   context = enc^T @ alpha streaming pass.
// b_full is omitted: softmax(x + c) == softmax(x) and raw scores are not
// outputs. Workspace: 2 MB bf16 prepacked W_enc in B-fragment order.
// ---------------------------------------------------------------------------

typedef __attribute__((ext_vector_type(16))) __bf16 v16bf;
typedef __attribute__((ext_vector_type(8)))  float  v8f;

#define ENC_DIM   2048
#define DEC_DIM   512
#define ATT_DIM   512
#define BATCH     256
#define NPIX      196
#define NPIX_PAD  224      // 7 M-passes * 32 rows
#define NTILES    32       // ATT_DIM / 16
#define KSTEPS    64       // ENC_DIM / 32

static __device__ __forceinline__ unsigned short f2bf(float f) {
  union { float f; unsigned u; } v; v.f = f;
  unsigned r = v.u + 0x7FFFu + ((v.u >> 16) & 1u);   // round-to-nearest-even
  return (unsigned short)(r >> 16);
}

// ---------------------------------------------------------------------------
// Prepack W_enc (2048x512 f32, row-major K x A) into bf16 B-fragment order:
//   offset_ushorts = ((nt*64 + Sg)*32 + lane)*16 + j
// lane holds column n = nt*16 + (lane&15); K = Sg*32 + (lane<16?0:16) + j.
// 65536 threads total.
// ---------------------------------------------------------------------------
__global__ __launch_bounds__(256)
void pack_Wenc(const float* __restrict__ W, unsigned short* __restrict__ dst) {
  int t    = blockIdx.x * 256 + threadIdx.x;
  int nt   = t >> 11;           // 0..31
  int rest = t & 2047;
  int Sg   = rest >> 5;         // 0..63
  int l    = rest & 31;
  int n     = nt * 16 + (l & 15);
  int kbase = Sg * 32 + ((l & 16) ? 16 : 0);
  unsigned short* o = dst + ((((nt * 64) + Sg) * 32 + l) << 4);
#pragma unroll
  for (int j = 0; j < 16; ++j)
    o[j] = f2bf(W[(size_t)(kbase + j) * ATT_DIM + n]);
}

// ---------------------------------------------------------------------------
// Main kernel: one block (256 threads = 8 wave32) per batch element.
// ---------------------------------------------------------------------------
__global__ __launch_bounds__(256)
void attn_kernel(const float* __restrict__ enc,
                 const float* __restrict__ dh,
                 const float* __restrict__ b_enc,
                 const float* __restrict__ W_dec,
                 const float* __restrict__ b_dec,
                 const float* __restrict__ w_full,
                 const unsigned short* __restrict__ packW,
                 float* __restrict__ out) {
  __shared__ float s_dh[DEC_DIM];
  __shared__ float s_att2[ATT_DIM];           // att2 + b_dec + b_enc
  __shared__ float s_wf[ATT_DIM];
  __shared__ float s_score[NPIX_PAD];         // scores, then alpha
  __shared__ float s_red[256];
  __shared__ __align__(32) unsigned short s_afrag[2 * 8 * 32 * 16]; // 16 KB

  const int t = threadIdx.x;
  const int b = blockIdx.x;
  const int l = t & 31;        // lane in wave
  const int w = t >> 5;        // wave id 0..7

  // ---------------- Phase A: att2 = dh @ W_dec + b_dec + b_enc -------------
  s_dh[t]       = dh[(size_t)b * DEC_DIM + t];
  s_dh[t + 256] = dh[(size_t)b * DEC_DIM + t + 256];
  s_wf[t]       = w_full[t];
  s_wf[t + 256] = w_full[t + 256];
  if (t < NPIX_PAD) s_score[t] = 0.f;
  __syncthreads();

  float a0acc = b_dec[t]       + b_enc[t];
  float a1acc = b_dec[t + 256] + b_enc[t + 256];
  for (int k = 0; k < DEC_DIM; ++k) {
    float h = s_dh[k];
    const float* Wr = W_dec + (size_t)k * ATT_DIM;
    a0acc += h * Wr[t];
    a1acc += h * Wr[t + 256];
  }
  s_att2[t]       = a0acc;
  s_att2[t + 256] = a1acc;
  __syncthreads();

  // ---------------- Phase B: score[p] = relu(enc@W_enc + att2) . w_full ----
  const float* encB = enc + (size_t)b * NPIX * ENC_DIM;

  for (int mp = 0; mp < 7; ++mp) {          // 32 rows per pass
    v8f c[2][4];
#pragma unroll
    for (int mt = 0; mt < 2; ++mt)
#pragma unroll
      for (int q = 0; q < 4; ++q)
#pragma unroll
        for (int v = 0; v < 8; ++v) c[mt][q][v] = 0.f;

    for (int kc = 0; kc < 8; ++kc) {        // 256-wide K chunks
      // stage 32 rows x 256 K of enc as bf16 A-fragments into LDS
#pragma unroll
      for (int it = 0; it < 4; ++it) {
        int lin = t * 8 + it * 2048;        // bijective over 32x256, step 8
        int r   = lin >> 8;                 // 0..31
        int kk  = lin & 255;                // multiple of 8
        int row = mp * 32 + r;
        float4 x0 = {0.f, 0.f, 0.f, 0.f}, x1 = {0.f, 0.f, 0.f, 0.f};
        if (row < NPIX) {
          const float4* src =
              (const float4*)(encB + (size_t)row * ENC_DIM + kc * 256 + kk);
          x0 = src[0];
          x1 = src[1];
        }
        int s    = kk >> 5;
        int rem  = kk & 31;                 // 0,8,16,24
        int lane = ((rem & 16) ? 16 : 0) + (r & 15);
        int mt   = r >> 4;
        int base = (((mt * 8 + s) * 32 + lane) << 4) + ((rem & 8) ? 8 : 0);
        uint4 pk;
        pk.x = (unsigned)f2bf(x0.x) | ((unsigned)f2bf(x0.y) << 16);
        pk.y = (unsigned)f2bf(x0.z) | ((unsigned)f2bf(x0.w) << 16);
        pk.z = (unsigned)f2bf(x1.x) | ((unsigned)f2bf(x1.y) << 16);
        pk.w = (unsigned)f2bf(x1.z) | ((unsigned)f2bf(x1.w) << 16);
        *reinterpret_cast<uint4*>(&s_afrag[base]) = pk;
      }
      __syncthreads();

#pragma unroll
      for (int s = 0; s < 8; ++s) {
        v16bf a0 = *reinterpret_cast<const v16bf*>(&s_afrag[((0 * 8 + s) * 32 + l) << 4]);
        v16bf a1 = *reinterpret_cast<const v16bf*>(&s_afrag[((1 * 8 + s) * 32 + l) << 4]);
        int Sg = kc * 8 + s;
#pragma unroll
        for (int q = 0; q < 4; ++q) {
          int nt = w * 4 + q;
          v16bf bf = *reinterpret_cast<const v16bf*>(
              packW + ((((nt * 64) + Sg) * 32 + l) << 4));
          c[0][q] = __builtin_amdgcn_wmma_f32_16x16x32_bf16(
              false, a0, false, bf, (short)0, c[0][q], false, false);
          c[1][q] = __builtin_amdgcn_wmma_f32_16x16x32_bf16(
              false, a1, false, bf, (short)0, c[1][q], false, false);
        }
      }
      __syncthreads();
    }

    // epilogue: +att2, relu, *w_full, reduce over N into per-row score
    float part[2][8];
#pragma unroll
    for (int mt = 0; mt < 2; ++mt)
#pragma unroll
      for (int v = 0; v < 8; ++v) part[mt][v] = 0.f;

#pragma unroll
    for (int q = 0; q < 4; ++q) {
      int a    = (w * 4 + q) * 16 + (l & 15);
      float t2 = s_att2[a];
      float wf = s_wf[a];
#pragma unroll
      for (int mt = 0; mt < 2; ++mt)
#pragma unroll
        for (int v = 0; v < 8; ++v) {
          float x = c[mt][q][v] + t2;
          part[mt][v] += fmaxf(x, 0.f) * wf;
        }
    }
    int mofs = mp * 32 + ((l & 16) ? 8 : 0);   // C layout: lanes>=16 hold M=v+8
#pragma unroll
    for (int mt = 0; mt < 2; ++mt)
#pragma unroll
      for (int v = 0; v < 8; ++v)
        atomicAdd(&s_score[mofs + mt * 16 + v], part[mt][v]);
    __syncthreads();
  }

  // ---------------- Phase C: softmax over P ---------------------------------
  float sv = (t < NPIX) ? s_score[t] : -3.402823466e38f;
  s_red[t] = sv;
  __syncthreads();
#pragma unroll
  for (int off = 128; off > 0; off >>= 1) {
    if (t < off) s_red[t] = fmaxf(s_red[t], s_red[t + off]);
    __syncthreads();
  }
  float mx = s_red[0];
  __syncthreads();
  float ev = (t < NPIX) ? __expf(sv - mx) : 0.f;
  s_red[t] = ev;
  __syncthreads();
#pragma unroll
  for (int off = 128; off > 0; off >>= 1) {
    if (t < off) s_red[t] += s_red[t + off];
    __syncthreads();
  }
  float av = ev * (1.f / s_red[0]);
  if (t < NPIX) {
    s_score[t] = av;                                  // reuse as alpha
    out[(size_t)BATCH * ENC_DIM + (size_t)b * NPIX + t] = av;
  }
  __syncthreads();

  // ---------------- Phase D: context = sum_p alpha[p] * enc[b,p,:] ----------
  float4 acc0 = {0.f, 0.f, 0.f, 0.f};
  float4 acc1 = {0.f, 0.f, 0.f, 0.f};
  for (int p = 0; p < NPIX; ++p) {
    float ap = s_score[p];
    const float4* r = (const float4*)(encB + (size_t)p * ENC_DIM);
    float4 v0 = r[t];
    float4 v1 = r[256 + t];
    acc0.x += ap * v0.x; acc0.y += ap * v0.y; acc0.z += ap * v0.z; acc0.w += ap * v0.w;
    acc1.x += ap * v1.x; acc1.y += ap * v1.y; acc1.z += ap * v1.z; acc1.w += ap * v1.w;
  }
  float4* o = (float4*)(out + (size_t)b * ENC_DIM);
  o[t]       = acc0;
  o[256 + t] = acc1;
}

// ---------------------------------------------------------------------------
extern "C" void kernel_launch(void* const* d_in, const int* in_sizes, int n_in,
                              void* d_out, int out_size, void* d_ws, size_t ws_size,
                              hipStream_t stream) {
  (void)in_sizes; (void)n_in; (void)out_size; (void)ws_size;
  const float* enc    = (const float*)d_in[0];
  const float* dh     = (const float*)d_in[1];
  const float* W_enc  = (const float*)d_in[2];
  const float* b_enc  = (const float*)d_in[3];
  const float* W_dec  = (const float*)d_in[4];
  const float* b_dec  = (const float*)d_in[5];
  const float* w_full = (const float*)d_in[6];
  // d_in[7] (b_full): softmax is shift-invariant and scores are not emitted.

  unsigned short* packW = (unsigned short*)d_ws;   // 2 MB bf16 W_enc
  float* out = (float*)d_out;

  pack_Wenc<<<256, 256, 0, stream>>>(W_enc, packW);
  attn_kernel<<<BATCH, 256, 0, stream>>>(enc, dh, b_enc, W_dec, b_dec,
                                         w_full, packW, out);
}
